// AttentionHead_455266533655
// MI455X (gfx1250) — matrix operand order repaired and checked by reading, not compiled, not run on hardware
//
#include <hip/hip_runtime.h>
#include <hip/hip_bf16.h>

#define S_LEN 4096
#define BATCH 4
#define NE    64

typedef _Float16 h8  __attribute__((ext_vector_type(8)));
typedef _Float16 h16 __attribute__((ext_vector_type(16)));
typedef float    f8  __attribute__((ext_vector_type(8)));

// scores are kept in the base-2 domain: scale = (1/sqrt(64)) * log2(e)
#define SCALE2 0.18033688011112042f

// Assemble a 16x32 f16 A-operand (v16h) for this lane from two 16B chunks:
// halves[0..7]  = p[0..7]   (kappa 0..7 for lanes<16 / 16..23 for lanes>=16)
// halves[8..15] = p[8..15]  (kappa 8..15 / 24..31)
static __device__ __forceinline__ h16 load_a_tile(const _Float16* p) {
  h8 lo = *(const h8*)(p);
  h8 hi = *(const h8*)(p + 8);
  return __builtin_shufflevector(lo, hi, 0,1,2,3,4,5,6,7,8,9,10,11,12,13,14,15);
}

static __device__ __forceinline__ f8 wmma16(h16 a, h16 b, f8 c) {
  return __builtin_amdgcn_wmma_f32_16x16x32_f16(false, a, false, b, (short)0, c,
                                                false, false);
}

// ---------------------------------------------------------------------------
// Phase 1: K/Q/V projections (fp32 math, f16 output). V stored transposed
// (Vt[b][dim][s]) so the P*V WMMA B-operand is contiguous in memory.
// ---------------------------------------------------------------------------
__global__ __launch_bounds__(256) void
proj_kernel(const float* __restrict__ x,
            const float* __restrict__ Wk, const float* __restrict__ bk,
            const float* __restrict__ Wq, const float* __restrict__ bq,
            const float* __restrict__ Wv, const float* __restrict__ bv,
            _Float16* __restrict__ K, _Float16* __restrict__ Q,
            _Float16* __restrict__ Vt) {
  const int idx = blockIdx.x * 256 + threadIdx.x;   // [0, S*B*64)
  const int n  = idx & 63;
  const int sb = idx >> 6;                          // s*B + b
  const int b  = sb & 3;
  const int s  = sb >> 2;
  const float* xr = x + sb * 5;
  const float x0 = xr[0], x1 = xr[1], x2 = xr[2], x3 = xr[3], x4 = xr[4];

  float k = bk[n] + x0*Wk[n] + x1*Wk[64+n] + x2*Wk[128+n] + x3*Wk[192+n] + x4*Wk[256+n];
  float q = bq[n] + x0*Wq[n] + x1*Wq[64+n] + x2*Wq[128+n] + x3*Wq[192+n] + x4*Wq[256+n];
  float v = bv[n] + x0*Wv[n] + x1*Wv[64+n] + x2*Wv[128+n] + x3*Wv[192+n] + x4*Wv[256+n];

  K[(size_t)(b * S_LEN + s) * NE + n] = (_Float16)k;
  Q[(size_t)(b * S_LEN + s) * NE + n] = (_Float16)q;
  Vt[(size_t)(b * NE + n) * S_LEN + s] = (_Float16)v;
}

// ---------------------------------------------------------------------------
// Phase 2: flash attention. One wave per 16-query tile; each wave processes
// tile g and tile 1023-g so all workgroups have equal work under causality.
// Inner loop consumes 32 keys per iteration: 4 score WMMAs + 4 PV WMMAs.
// K operands are double-buffered across iterations and V operands issued at
// the top of the iteration, so the softmax VALU chain hides load latency.
// ---------------------------------------------------------------------------
__global__ __launch_bounds__(128) void
attn_kernel(const _Float16* __restrict__ Kg, const _Float16* __restrict__ Qg,
            const _Float16* __restrict__ Vt, float* __restrict__ out) {
  __shared__ _Float16 Plds[4][16 * 32];   // per-wave P staging tile (1 KB each)

  const int wave = threadIdx.x >> 5;
  const int lane = threadIdx.x & 31;
  const int l16  = lane & 15;
  const int hh   = lane >> 4;             // lane half
  _Float16* pb = Plds[wave];
  const int g = blockIdx.x * 4 + wave;    // 0..511

#pragma unroll 1
  for (int pick = 0; pick < 2; ++pick) {
    const int t     = pick ? (1023 - g) : g;
    const int b     = t >> 8;
    const int qtile = t & 255;
    const int q0    = qtile << 4;

    const _Float16* Kb = Kg + (size_t)b * S_LEN * NE;
    const _Float16* Qb = Qg + (size_t)b * S_LEN * NE;
    const _Float16* Vb = Vt + (size_t)b * NE * S_LEN;

    // Q tile as two A-operands (head dims 0..31 and 32..63), loop-resident
    const _Float16* qp = Qb + (size_t)(q0 + l16) * NE + hh * 16;
    const h16 aQ0 = load_a_tile(qp);
    const h16 aQ1 = load_a_tile(qp + 32);

    f8 O0 = {}, O1 = {}, O2 = {}, O3 = {};
    float m[8], l[8];
#pragma unroll
    for (int r = 0; r < 8; ++r) { m[r] = -__builtin_inff(); l[r] = 0.0f; }

    const int ngrp = (qtile >> 1) + 1;    // 32-key groups covering keys 0..q0+15

    // prologue: K operands for group 0
    const _Float16* kp0 = Kb + (size_t)l16 * NE + hh * 16;
    h16 bK0a = *(const h16*)(kp0);                 // keys 0..15,  dims 0..31
    h16 bK1a = *(const h16*)(kp0 + 32);            // keys 0..15,  dims 32..63
    h16 bK0b = *(const h16*)(kp0 + 16 * NE);       // keys 16..31, dims 0..31
    h16 bK1b = *(const h16*)(kp0 + 16 * NE + 32);  // keys 16..31, dims 32..63

#pragma unroll 1
    for (int grp = 0; grp < ngrp; ++grp) {
      const int k0 = grp << 5;

      // V operands for this group, issued early so softmax hides their latency
      const _Float16* vbase = Vb + k0 + hh * 16;
      const h16 bV0 = *(const h16*)(vbase + (size_t)(l16)      * S_LEN);
      const h16 bV1 = *(const h16*)(vbase + (size_t)(16 + l16) * S_LEN);
      const h16 bV2 = *(const h16*)(vbase + (size_t)(32 + l16) * S_LEN);
      const h16 bV3 = *(const h16*)(vbase + (size_t)(48 + l16) * S_LEN);

      // scores: two independent accumulator chains, interleaved
      f8 c1 = {}, c2 = {};
      c1 = wmma16(aQ0, bK0a, c1);
      c2 = wmma16(aQ0, bK0b, c2);
      c1 = wmma16(aQ1, bK1a, c1);
      c2 = wmma16(aQ1, bK1b, c2);

      // double-buffer: start next group's K loads now (regs just consumed)
      if (grp + 1 < ngrp) {
        const _Float16* kp = Kb + (size_t)(k0 + 32 + l16) * NE + hh * 16;
        __builtin_prefetch(kp + 32 * NE, 0, 1);    // prefetch group grp+2
        bK0a = *(const h16*)(kp);
        bK1a = *(const h16*)(kp + 32);
        bK0b = *(const h16*)(kp + 16 * NE);
        bK1b = *(const h16*)(kp + 16 * NE + 32);
      }

      float s1[8], s2[8];
#pragma unroll
      for (int r = 0; r < 8; ++r) { s1[r] = c1[r] * SCALE2; s2[r] = c2[r] * SCALE2; }

      if (grp == ngrp - 1) {              // group touching the diagonal: mask key>query
#pragma unroll
        for (int r = 0; r < 8; ++r) {
          const int qrow = q0 + r + 8 * hh;
          if (k0 + l16      > qrow) s1[r] = -__builtin_inff();
          if (k0 + 16 + l16 > qrow) s2[r] = -__builtin_inff();
        }
      }

      // online softmax per query row; reductions shared across both key tiles
      float av[8];
#pragma unroll
      for (int r = 0; r < 8; ++r) {
        float rm = fmaxf(s1[r], s2[r]);
        rm = fmaxf(rm, __shfl_xor(rm, 1, 32));
        rm = fmaxf(rm, __shfl_xor(rm, 2, 32));
        rm = fmaxf(rm, __shfl_xor(rm, 4, 32));
        rm = fmaxf(rm, __shfl_xor(rm, 8, 32));
        const float mn    = fmaxf(m[r], rm);
        const float alpha = exp2f(m[r] - mn);
        const float p1    = exp2f(s1[r] - mn);
        const float p2    = exp2f(s2[r] - mn);
        float rs = p1 + p2;
        rs += __shfl_xor(rs, 1, 32);
        rs += __shfl_xor(rs, 2, 32);
        rs += __shfl_xor(rs, 4, 32);
        rs += __shfl_xor(rs, 8, 32);
        l[r] = l[r] * alpha + rs;
        m[r] = mn;
        av[r] = alpha;
        s1[r] = p1;
        s2[r] = p2;
      }
      // rescale output accumulators (same M residency as av[])
#pragma unroll
      for (int r = 0; r < 8; ++r) {
        O0[r] *= av[r]; O1[r] *= av[r]; O2[r] *= av[r]; O3[r] *= av[r];
      }
      // stash the 16x32 P tile into LDS (row-major: row = query, col = local key)
#pragma unroll
      for (int r = 0; r < 8; ++r) {
        const int rowoff = (r + 8 * hh) * 32;
        pb[rowoff + l16]      = (_Float16)s1[r];
        pb[rowoff + 16 + l16] = (_Float16)s2[r];
      }
      asm volatile("s_wait_dscnt 0" ::: "memory");
      // reload P in A-operand layout (16q x 32k) and do the PV WMMAs
      const h16 aP = load_a_tile(pb + l16 * 32 + hh * 16);
      O0 = wmma16(aP, bV0, O0);
      O1 = wmma16(aP, bV1, O1);
      O2 = wmma16(aP, bV2, O2);
      O3 = wmma16(aP, bV3, O3);
    }

    // epilogue: normalize and store. out[(q*B + b)*64 + e], e = t*16 + l16
#pragma unroll
    for (int r = 0; r < 8; ++r) {
      const float inv = 1.0f / l[r];
      const int row = q0 + r + 8 * hh;
      float* op = out + ((size_t)row * BATCH + b) * NE + l16;
      op[0]  = O0[r] * inv;
      op[16] = O1[r] * inv;
      op[32] = O2[r] * inv;
      op[48] = O3[r] * inv;
    }
  }
}

// ---------------------------------------------------------------------------
extern "C" void kernel_launch(void* const* d_in, const int* in_sizes, int n_in,
                              void* d_out, int out_size, void* d_ws, size_t ws_size,
                              hipStream_t stream) {
  const float* x  = (const float*)d_in[0];
  const float* Wk = (const float*)d_in[1];
  const float* bk = (const float*)d_in[2];
  const float* Wq = (const float*)d_in[3];
  const float* bq = (const float*)d_in[4];
  const float* Wv = (const float*)d_in[5];
  const float* bv = (const float*)d_in[6];

  // workspace: K | Q | Vt, each B*S*64 f16 (2 MB) -> 6 MB total
  _Float16* Kw  = (_Float16*)d_ws;
  _Float16* Qw  = Kw + (size_t)BATCH * S_LEN * NE;
  _Float16* Vtw = Qw + (size_t)BATCH * S_LEN * NE;

  proj_kernel<<<(S_LEN * BATCH * NE) / 256, 256, 0, stream>>>(
      x, Wk, bk, Wq, bq, Wv, bv, Kw, Qw, Vtw);

  attn_kernel<<<128, 128, 0, stream>>>(Kw, Qw, Vtw, (float*)d_out);
}
